// SNN_59742995087749
// MI455X (gfx1250) — matrix operand order repaired and checked by reading, not compiled, not run on hardware
//
#include <hip/hip_runtime.h>

typedef __attribute__((ext_vector_type(16))) _Float16 v16h;
typedef __attribute__((ext_vector_type(8)))  _Float16 v8h;
typedef __attribute__((ext_vector_type(8)))  float    v8f;
typedef __attribute__((ext_vector_type(4)))  float    v4f;

#define T_STEPS 64
#define BATCH   8192
#define DIN     784
#define HID     20
#define NOUT    10
#define KPAD    800      // 25 chunks of K=32, zero-padded past 784
#define NCHUNK  25
#define ROWS_PER_WAVE   16
#define WAVES_PER_BLOCK 4
#define ROWS_PER_BLOCK  (ROWS_PER_WAVE * WAVES_PER_BLOCK)
#define BTILES          (BATCH / 16)                 // 512 batch tiles
#define TILES_TOTAL     (T_STEPS * BTILES)           // 32768 M-tiles (t-major)
#define P1_TILES_PER_WAVE 8
#define SCRATCH_HALFS_PER_TILE 512                   // 32 lanes x 16 halfs (C-layout dump)

// ---------------- shared device helpers ----------------

__device__ __forceinline__ void stage_w1(const float* __restrict__ w1,
                                         _Float16* __restrict__ ldsw,
                                         int tid, int nthreads) {
    for (int i = tid; i < 32 * KPAD; i += nthreads) {
        const int h = i / KPAD;
        const int k = i - h * KPAD;
        const float v = (h < HID && k < DIN) ? w1[h * DIN + k] : 0.0f;
        ldsw[i] = (_Float16)v;
    }
}

// One 16x784 x 784x32 tile of fc1: x rows -> A fragments, w1 (LDS) -> B fragments.
__device__ __forceinline__ void fc1_accumulate(const float* __restrict__ xb,
                                               const _Float16* __restrict__ ldsw,
                                               int nIdx, int koff, int kcon,
                                               const float* __restrict__ pfb,
                                               v8f& acc0, v8f& acc1) {
    #pragma unroll 5
    for (int c = 0; c < NCHUNK; ++c) {
        const int k0 = c * 32;
        __builtin_prefetch(pfb + k0 + koff, 0, 0);   // speculative; dropped if OOB
        v16h a;
        {
            const v4f g0 = *(const v4f*)(xb + k0 + koff);
            const v4f g1 = *(const v4f*)(xb + k0 + koff + 4);
            #pragma unroll
            for (int i = 0; i < 4; ++i) { a[i] = (_Float16)g0[i]; a[4 + i] = (_Float16)g1[i]; }
        }
        if (c < NCHUNK - 1) {
            const v4f g2 = *(const v4f*)(xb + k0 + 16 + koff);
            const v4f g3 = *(const v4f*)(xb + k0 + 16 + koff + 4);
            #pragma unroll
            for (int i = 0; i < 4; ++i) { a[8 + i] = (_Float16)g2[i]; a[12 + i] = (_Float16)g3[i]; }
        } else {
            #pragma unroll
            for (int i = 0; i < 8; ++i) a[8 + i] = (_Float16)0.0f;    // K 784..799 pad
        }
        const _Float16* w0p = ldsw + nIdx * KPAD + k0 + koff;
        const _Float16* w1p = ldsw + (nIdx + 16) * KPAD + k0 + koff;
        const v8h bl0 = *(const v8h*)(w0p);
        const v8h bh0 = *(const v8h*)(w0p + 16);
        const v8h bl1 = *(const v8h*)(w1p);
        const v8h bh1 = *(const v8h*)(w1p + 16);
        v16h bf0, bf1;
        #pragma unroll
        for (int i = 0; i < 8; ++i) {
            bf0[i] = bl0[i]; bf0[8 + i] = bh0[i];
            bf1[i] = bl1[i]; bf1[8 + i] = bh1[i];
        }
        acc0 = __builtin_amdgcn_wmma_f32_16x16x32_f16(false, a, false, bf0, (short)0, acc0, false, false);
        acc1 = __builtin_amdgcn_wmma_f32_16x16x32_f16(false, a, false, bf1, (short)0, acc1, false, false);
    }
}

// IF layer-1 update + s1 transpose via wave-private LDS + fc2 WMMA + IF layer-2.
__device__ __forceinline__ void if_scan_step(const v8f& acc0, const v8f& acc1,
                                             v8f& v1a, v8f& v1b, v8f& v2,
                                             float bias1a, float bias1b, float bias2,
                                             _Float16* __restrict__ myS1, const v16h& w2f,
                                             int nIdx, int half, int koff,
                                             bool lastStep, int b0, float* __restrict__ out) {
    #pragma unroll
    for (int j = 0; j < 8; ++j) {
        const int m = half * 8 + j;
        const float nva = v1a[j] + acc0[j] + bias1a;
        const float sa  = (nva >= 1.0f) ? 1.0f : 0.0f;   // spike(v - V_TH)
        v1a[j] = nva * (1.0f - sa);                       // hard reset
        myS1[m * 32 + nIdx] = (_Float16)sa;
        const float nvb = v1b[j] + acc1[j] + bias1b;
        const float sb  = (nvb >= 1.0f) ? 1.0f : 0.0f;
        v1b[j] = nvb * (1.0f - sb);
        myS1[m * 32 + nIdx + 16] = (_Float16)sb;
    }
    __builtin_amdgcn_wave_barrier();   // wave-private tile; DS is in-order within a wave
    v16h sfrag;
    {
        const _Float16* sp = myS1 + nIdx * 32;
        const v8h sA = *(const v8h*)(sp + koff);
        const v8h sB = *(const v8h*)(sp + 16 + koff);
        #pragma unroll
        for (int i = 0; i < 8; ++i) { sfrag[i] = sA[i]; sfrag[8 + i] = sB[i]; }
    }
    v8f zc = {};
    const v8f acc2 = __builtin_amdgcn_wmma_f32_16x16x32_f16(false, sfrag, false, w2f, (short)0, zc, false, false);
    #pragma unroll
    for (int j = 0; j < 8; ++j) {
        const float nv2 = v2[j] + acc2[j] + bias2;
        const float s2  = (nv2 >= 1.0f) ? 1.0f : 0.0f;
        v2[j] = nv2 * (1.0f - s2);
        if (lastStep && nIdx < NOUT) {
            out[(size_t)(b0 + half * 8 + j) * NOUT + nIdx] = s2;
        }
    }
}

// fc2 constant B fragment: B[k=h][n=o] = w2[o][h], 32x16 f16.
__device__ __forceinline__ v16h make_w2_frag(const float* __restrict__ w2, int nIdx, int half) {
    v16h w2f;
    const int kbase = half * 8;
    #pragma unroll
    for (int i = 0; i < 16; ++i) {
        const int h = kbase + i + (i >= 8 ? 8 : 0);
        const float v = (nIdx < NOUT && h < HID) ? w2[nIdx * HID + h] : 0.0f;
        w2f[i] = (_Float16)v;
    }
    return w2f;
}

// ---------------- Phase 1: streaming fc1 GEMM (massively parallel) ----------------
// Grid covers all T*B/16 M-tiles; dumps C-layout accumulators verbatim as f16.
__global__ __launch_bounds__(WAVES_PER_BLOCK * 32)
void snn_fc1_gemm_kernel(const float* __restrict__ x, const float* __restrict__ w1,
                         _Float16* __restrict__ a1s) {
    __shared__ _Float16 lds_w[32 * KPAD];
    const int tid  = threadIdx.x;
    const int lane = tid & 31;
    const int wave = tid >> 5;
    const int nIdx = lane & 15;
    const int half = lane >> 4;
    stage_w1(w1, lds_w, tid, WAVES_PER_BLOCK * 32);
    __syncthreads();

    const int koff = half * 8, kcon = half * 16;
    const int tile0 = (blockIdx.x * WAVES_PER_BLOCK + wave) * P1_TILES_PER_WAVE;
    #pragma unroll 1
    for (int ti = 0; ti < P1_TILES_PER_WAVE; ++ti) {
        const int tile = tile0 + ti;                               // t-major: row = tile*16 + m
        const float* xb = x + ((size_t)tile * 16 + nIdx) * DIN;    // A row = nIdx (both halves)
        v8f acc0 = {}, acc1 = {};
        fc1_accumulate(xb, lds_w, nIdx, koff, kcon, xb + (size_t)16 * DIN, acc0, acc1);
        _Float16* dst = a1s + ((size_t)tile * 32 + lane) * 16;     // 32 B/lane, coalesced
        v8h o0, o1;
        #pragma unroll
        for (int j = 0; j < 8; ++j) { o0[j] = (_Float16)acc0[j]; o1[j] = (_Float16)acc1[j]; }
        *(v8h*)dst       = o0;
        *(v8h*)(dst + 8) = o1;
    }
}

// ---------------- Phase 2: IF scan + fc2 over the precomputed activations ----------------
__global__ __launch_bounds__(WAVES_PER_BLOCK * 32)
void snn_scan_kernel(const _Float16* __restrict__ a1s,
                     const float* __restrict__ b1, const float* __restrict__ w2,
                     const float* __restrict__ b2, float* __restrict__ out) {
    __shared__ _Float16 lds_s1[WAVES_PER_BLOCK][ROWS_PER_WAVE * 32];
    const int tid  = threadIdx.x;
    const int lane = tid & 31;
    const int wave = tid >> 5;
    const int nIdx = lane & 15;
    const int half = lane >> 4;
    const int koff = half * 8;
    const int btile = blockIdx.x * WAVES_PER_BLOCK + wave;   // 0..511
    const int b0 = btile * 16;

    const float bias1a = (nIdx < HID)        ? b1[nIdx]      : 0.0f;
    const float bias1b = ((nIdx + 16) < HID) ? b1[nIdx + 16] : 0.0f;
    const float bias2  = (nIdx < NOUT)       ? b2[nIdx]      : 0.0f;
    const v16h w2f = make_w2_frag(w2, nIdx, half);

    v8f v1a = {}, v1b = {}, v2 = {};
    _Float16* const myS1 = &lds_s1[wave][0];
    const _Float16* src = a1s + ((size_t)btile * 32 + lane) * 16;
    const size_t tstride = (size_t)BTILES * SCRATCH_HALFS_PER_TILE;  // halfs per timestep

    for (int t = 0; t < T_STEPS; ++t) {
        const _Float16* p = src + (size_t)t * tstride;
        __builtin_prefetch(p + tstride, 0, 0);
        const v8h r0 = *(const v8h*)(p);
        const v8h r1 = *(const v8h*)(p + 8);
        v8f acc0, acc1;
        #pragma unroll
        for (int j = 0; j < 8; ++j) { acc0[j] = (float)r0[j]; acc1[j] = (float)r1[j]; }
        if_scan_step(acc0, acc1, v1a, v1b, v2, bias1a, bias1b, bias2,
                     myS1, w2f, nIdx, half, koff, t == T_STEPS - 1, b0, out);
    }
}

// ---------------- Fused fallback (small-workspace path) ----------------
__global__ __launch_bounds__(WAVES_PER_BLOCK * 32)
void snn_if_mlp_fused_kernel(const float* __restrict__ x, const float* __restrict__ w1,
                             const float* __restrict__ b1, const float* __restrict__ w2,
                             const float* __restrict__ b2, float* __restrict__ out) {
    __shared__ _Float16 lds_w[32 * KPAD];
    __shared__ _Float16 lds_s1[WAVES_PER_BLOCK][ROWS_PER_WAVE * 32];
    const int tid  = threadIdx.x;
    const int lane = tid & 31;
    const int wave = tid >> 5;
    const int nIdx = lane & 15;
    const int half = lane >> 4;
    stage_w1(w1, lds_w, tid, WAVES_PER_BLOCK * 32);
    __syncthreads();

    const int b0 = blockIdx.x * ROWS_PER_BLOCK + wave * ROWS_PER_WAVE;
    const float bias1a = (nIdx < HID)        ? b1[nIdx]      : 0.0f;
    const float bias1b = ((nIdx + 16) < HID) ? b1[nIdx + 16] : 0.0f;
    const float bias2  = (nIdx < NOUT)       ? b2[nIdx]      : 0.0f;
    const v16h w2f = make_w2_frag(w2, nIdx, half);

    v8f v1a = {}, v1b = {}, v2 = {};
    _Float16* const myS1 = &lds_s1[wave][0];
    const int koff = half * 8, kcon = half * 16;
    const float* const xlane = x + (size_t)(b0 + nIdx) * DIN;
    const size_t tstride = (size_t)BATCH * DIN;

    for (int t = 0; t < T_STEPS; ++t) {
        const float* xb = xlane + (size_t)t * tstride;
        v8f acc0 = {}, acc1 = {};
        fc1_accumulate(xb, lds_w, nIdx, koff, kcon, xb + tstride, acc0, acc1);
        if_scan_step(acc0, acc1, v1a, v1b, v2, bias1a, bias1b, bias2,
                     myS1, w2f, nIdx, half, koff, t == T_STEPS - 1, b0, out);
    }
}

// ---------------- host ----------------
extern "C" void kernel_launch(void* const* d_in, const int* in_sizes, int n_in,
                              void* d_out, int out_size, void* d_ws, size_t ws_size,
                              hipStream_t stream) {
    (void)in_sizes; (void)n_in; (void)out_size;
    const float* x  = (const float*)d_in[0];
    const float* w1 = (const float*)d_in[1];
    const float* b1 = (const float*)d_in[2];
    const float* w2 = (const float*)d_in[3];
    const float* b2 = (const float*)d_in[4];
    float* out = (float*)d_out;

    const size_t scratch_need = (size_t)TILES_TOTAL * SCRATCH_HALFS_PER_TILE * 2;  // 32 MB f16
    if (d_ws != nullptr && ws_size >= scratch_need) {
        // Decoupled path: streaming GEMM at full parallelism, then cheap scan.
        _Float16* a1s = (_Float16*)d_ws;
        dim3 g1(TILES_TOTAL / (WAVES_PER_BLOCK * P1_TILES_PER_WAVE));  // 1024 WGs
        dim3 blk(WAVES_PER_BLOCK * 32);                                // 128 thr = 4 waves
        snn_fc1_gemm_kernel<<<g1, blk, 0, stream>>>(x, w1, a1s);
        dim3 g2(BTILES / WAVES_PER_BLOCK);                             // 128 WGs
        snn_scan_kernel<<<g2, blk, 0, stream>>>(a1s, b1, w2, b2, out);
    } else {
        dim3 grid(BATCH / ROWS_PER_BLOCK);   // 128 WGs
        dim3 blk(WAVES_PER_BLOCK * 32);
        snn_if_mlp_fused_kernel<<<grid, blk, 0, stream>>>(x, w1, b1, w2, b2, out);
    }
}
